// CoMPILE_45964740002519
// MI455X (gfx1250) — matrix operand order repaired and checked by reading, not compiled, not run on hardware
//
#include <hip/hip_runtime.h>
#include <hip/hip_bf16.h>
#include <math.h>

#define kB    128
#define kNPG  400
#define kEPG  800
#define kN    (kB * kNPG)   /* 51200 nodes  */
#define kE    (kB * kEPG)   /* 102400 edges */
#define kH    128

typedef __attribute__((ext_vector_type(16))) __bf16   v16bf;
typedef __attribute__((ext_vector_type(8)))  float    v8f;
typedef __attribute__((ext_vector_type(8)))  unsigned v8u;
typedef __attribute__((ext_vector_type(4)))  float    v4f;

// pack two f32 into one dword of two bf16 (round-to-nearest-even)
__device__ __forceinline__ unsigned pack2bf(float x0, float x1) {
  unsigned u0 = __builtin_bit_cast(unsigned, x0);
  unsigned u1 = __builtin_bit_cast(unsigned, x1);
  unsigned r0 = u0 + 0x7FFFu + ((u0 >> 16) & 1u);
  unsigned r1 = u1 + 0x7FFFu + ((u1 >> 16) & 1u);
  return (r0 >> 16) | (r1 & 0xFFFF0000u);
}
__device__ __forceinline__ __bf16 f2bf(float x) {
  unsigned u = __builtin_bit_cast(unsigned, x);
  unsigned r = u + 0x7FFFu + ((u >> 16) & 1u);
  unsigned short h = (unsigned short)(r >> 16);
  return __builtin_bit_cast(__bf16, h);
}
// native-transcendental activations (v_exp_f32 / v_rcp_f32)
__device__ __forceinline__ float sigm(float x) {
  return __builtin_amdgcn_rcpf(1.f + __expf(-x));
}
__device__ __forceinline__ float tanh_fast(float x) {
  return 2.f * sigm(2.f * x) - 1.f;
}

// A fragment (16x32 bf16): lane half h covers K = 16h+0..15 contiguous
// relative to k0 -> two contiguous 8-float runs -> 4x b128 loads + pack.
__device__ __forceinline__ v16bf pack8(v4f a0, v4f a1, v4f b0, v4f b1) {
  v8u u;
  u[0] = pack2bf(a0[0], a0[1]); u[1] = pack2bf(a0[2], a0[3]);
  u[2] = pack2bf(a1[0], a1[1]); u[3] = pack2bf(a1[2], a1[3]);
  u[4] = pack2bf(b0[0], b0[1]); u[5] = pack2bf(b0[2], b0[3]);
  u[6] = pack2bf(b1[0], b1[1]); u[7] = pack2bf(b1[2], b1[3]);
  return __builtin_bit_cast(v16bf, u);
}
__device__ __forceinline__ v16bf afrag_plain(const float* p, int half) {
  const v4f* q0 = (const v4f*)(p + 16 * half);
  const v4f* q1 = (const v4f*)(p + 16 * half + 8);
  return pack8(q0[0], q0[1], q1[0], q1[1]);
}
__device__ __forceinline__ v16bf afrag_triple(const float* pa, const float* pb,
                                              const float* pc, int half) {
  int o0 = 16 * half, o1 = 16 * half + 8;
  v4f a0 = *(const v4f*)(pa + o0)     + *(const v4f*)(pb + o0)     - *(const v4f*)(pc + o0);
  v4f a1 = *(const v4f*)(pa + o0 + 4) + *(const v4f*)(pb + o0 + 4) - *(const v4f*)(pc + o0 + 4);
  v4f b0 = *(const v4f*)(pa + o1)     + *(const v4f*)(pb + o1)     - *(const v4f*)(pc + o1);
  v4f b1 = *(const v4f*)(pa + o1 + 4) + *(const v4f*)(pb + o1 + 4) - *(const v4f*)(pc + o1 + 4);
  return pack8(a0, a1, b0, b1);
}
// B fragment (32x16 bf16): lane half h holds K = 16h+0..15 contiguous -> one 32B load.
__device__ __forceinline__ v16bf bfrag(const __bf16* p, int half) {
  return *(const v16bf*)(p + 16 * half);
}

// ---------------------------------------------------------------- generic WMMA GEMM
// out[M,Nc] = act( Aop[M,K] @ Wt[Nc,K]^T + bias + resid ), A fused per MODE.
// Each wave computes TWO 16x16 N-tiles (n0, n0+16), reusing every A fragment twice.
struct GemmArgs {
  int mode, M, K, Nc, lda, ldr, ldo, act;
  const float* A;
  const float* nodef;   // node_feat or message_node
  const float* rel;     // final_rel [NREL,128]
  const float* trh;     // tgt_rel_h [E,128]
  const float* medge;   // message_edge [E,128]
  const int* esrc; const int* edst; const int* etype; const int* glabel;
  const int* spe; const int* dpe;
  const __bf16* W;      // Wt bf16 [Nc][K]
  const float* bias;
  const float* resid;
  float* out;
};

template <int MODE>
__global__ __launch_bounds__(256) void gemm_k(GemmArgs g) {
  const int wave = threadIdx.x >> 5, lane = threadIdx.x & 31;
  const int r = lane & 15, half = lane >> 4;
  const int m0 = blockIdx.x * 128 + wave * 16;
  const int n0 = blockIdx.y * 32;
  const int row = m0 + r;
  const __bf16* wrow0 = g.W + (size_t)(n0 + r) * g.K;
  const __bf16* wrow1 = g.W + (size_t)(n0 + 16 + r) * g.K;
  v8f acc0 = {0.f, 0.f, 0.f, 0.f, 0.f, 0.f, 0.f, 0.f};
  v8f acc1 = acc0;

#define WMMA_STEP(AF, KOFF)                                                          \
  do {                                                                               \
    v16bf _a = (AF);                                                                 \
    acc0 = __builtin_amdgcn_wmma_f32_16x16x32_bf16(false, _a, false,                 \
               bfrag(wrow0 + (KOFF), half), (short)0, acc0, false, false);           \
    acc1 = __builtin_amdgcn_wmma_f32_16x16x32_bf16(false, _a, false,                 \
               bfrag(wrow1 + (KOFF), half), (short)0, acc1, false, false);           \
  } while (0)

  if constexpr (MODE == 0) {            // plain rows of A, K in {128,256,384}
    const float* p0 = g.A + (size_t)row * g.lda;
    for (int k = 0; k < g.K; k += 32) WMMA_STEP(afrag_plain(p0 + k, half), k);
  } else if constexpr (MODE == 2) {     // rel[glabel[graph(e)]], K=128
    const float* p0 = g.rel + (size_t)g.glabel[row / kEPG] * 128;
#pragma unroll
    for (int k = 0; k < 128; k += 32) WMMA_STEP(afrag_plain(p0 + k, half), k);
  } else if constexpr (MODE == 1) {     // [nf[src] | rel[type] | nf[dst]], K=384
    const float* p0 = g.nodef + (size_t)g.esrc[row] * 128;
    const float* p1 = g.rel   + (size_t)g.etype[row] * 128;
    const float* p2 = g.nodef + (size_t)g.edst[row] * 128;
#pragma unroll
    for (int k = 0; k < 128; k += 32) WMMA_STEP(afrag_plain(p0 + k, half), k);
#pragma unroll
    for (int k = 0; k < 128; k += 32) WMMA_STEP(afrag_plain(p1 + k, half), 128 + k);
#pragma unroll
    for (int k = 0; k < 128; k += 32) WMMA_STEP(afrag_plain(p2 + k, half), 256 + k);
  } else if constexpr (MODE == 3) {     // mn[src] + trh - mn[dst], K=128
    const float* p0 = g.nodef + (size_t)g.esrc[row] * 128;
    const float* p1 = g.trh + (size_t)row * 128;
    const float* p2 = g.nodef + (size_t)g.edst[row] * 128;
#pragma unroll
    for (int k = 0; k < 128; k += 32)
      WMMA_STEP(afrag_triple(p0 + k, p1 + k, p2 + k, half), k);
  } else if constexpr (MODE == 4) {     // [mn[spe]+trh-mn[dpe] | message_edge], K=256
    int gr = row / kEPG;
    const float* p0 = g.nodef + (size_t)g.spe[gr] * 128;
    const float* p1 = g.trh + (size_t)row * 128;
    const float* p2 = g.nodef + (size_t)g.dpe[gr] * 128;
    const float* pm = g.medge + (size_t)row * 128;
#pragma unroll
    for (int k = 0; k < 128; k += 32)
      WMMA_STEP(afrag_triple(p0 + k, p1 + k, p2 + k, half), k);
#pragma unroll
    for (int k = 0; k < 128; k += 32) WMMA_STEP(afrag_plain(pm + k, half), 128 + k);
  } else {                              // 5: [message_edge | rel[glabel]], K=256
    const float* p0 = g.medge + (size_t)row * 128;
    const float* p1 = g.rel + (size_t)g.glabel[row / kEPG] * 128;
#pragma unroll
    for (int k = 0; k < 128; k += 32) WMMA_STEP(afrag_plain(p0 + k, half), k);
#pragma unroll
    for (int k = 0; k < 128; k += 32) WMMA_STEP(afrag_plain(p1 + k, half), 128 + k);
  }
#undef WMMA_STEP

  const int col0 = n0 + (lane & 15);
  const int col1 = col0 + 16;
  const int rb = m0 + (lane >> 4) * 8;
  const float bv0 = g.bias ? g.bias[col0] : 0.f;
  const float bv1 = g.bias ? g.bias[col1] : 0.f;
#pragma unroll
  for (int v = 0; v < 8; ++v) {
    float val0 = acc0[v] + bv0;
    float val1 = acc1[v] + bv1;
    if (g.resid) {
      val0 += g.resid[(size_t)(rb + v) * g.ldr + col0];
      val1 += g.resid[(size_t)(rb + v) * g.ldr + col1];
    }
    if (g.act) { val0 = fmaxf(val0, 0.f); val1 = fmaxf(val1, 0.f); }
    g.out[(size_t)(rb + v) * g.ldo + col0] = val0;
    g.out[(size_t)(rb + v) * g.ldo + col1] = val1;
  }
}

// ---------------------------------------------------------------- small kernels
__global__ void wconv_k(const float* src, __bf16* dst, int K, int Nc) {
  int i = blockIdx.x * 256 + threadIdx.x;
  if (i >= K * Nc) return;
  int n = i / K, k = i - n * K;
  dst[i] = f2bf(src[(size_t)k * Nc + n]);   // transpose: dst[Nc][K]
}
__global__ void attdot_k(const float* hid, const float* w, float* att) {
  int e = blockIdx.x * 256 + threadIdx.x;
  if (e >= kE) return;
  const v4f* h4 = (const v4f*)(hid + (size_t)e * 128);
  const v4f* w4 = (const v4f*)w;
  float s = 0.f;
#pragma unroll 8
  for (int k = 0; k < 32; ++k) {
    v4f a = h4[k], b = w4[k];
    s += a[0] * b[0] + a[1] * b[1] + a[2] * b[2] + a[3] * b[3];
  }
  att[e] = sigm(s);
}
__global__ void scale_k(const float* src, const float* att, float* dst) {
  size_t i = (size_t)blockIdx.x * 256 + threadIdx.x;   // E*128
  int e = (int)(i >> 7);
  dst[i] = src[i] * att[e];
}
__global__ void catprep_k(const float* mn, float* cat) {
  size_t i = (size_t)blockIdx.x * 256 + threadIdx.x;   // N*128
  int n = (int)(i >> 7), k = (int)(i & 127);
  cat[(size_t)n * 384 + k] = mn[i];
  cat[(size_t)n * 384 + 128 + k] = 0.f;
  cat[(size_t)n * 384 + 256 + k] = 0.f;
}
__global__ void scatter_k(const float* me, const int* edst, const int* esrc, float* cat) {
  int e = blockIdx.x, k = threadIdx.x;                 // grid=E, block=128
  float v = me[(size_t)e * 128 + k];
  atomicAdd(cat + (size_t)edst[e] * 384 + 128 + k, v);
  atomicAdd(cat + (size_t)esrc[e] * 384 + 256 + k, v);
}
__global__ void msg_k(const float* mn, const float* gb, float* msg) {
  size_t i = (size_t)blockIdx.x * 256 + threadIdx.x;   // N*128
  msg[i] = fmaxf(mn[i] + gb[i & 127], 0.f);
}
__global__ void h0max_k(const float* msg, float* h0) {
  int i = blockIdx.x * 256 + threadIdx.x;              // B*128
  int b = i >> 7, k = i & 127;
  float m = -1e30f;
  for (int t = 0; t < kNPG; ++t)
    m = fmaxf(m, msg[((size_t)(b * kNPG + t)) * 128 + k]);
  h0[i] = m;
}
__global__ void head_k(const float* nh, const float* rel, const int* sn, const int* tn,
                       const int* gl, const float* l1W, const float* l1b,
                       const float* l2W, const float* l2b, float* out) {
  int b = threadIdx.x;                                 // block=128
  const float* s = nh + (size_t)sn[b] * 128;
  const float* t = nh + (size_t)tn[b] * 128;
  const float* r = rel + (size_t)gl[b] * 128;
  float conv[128];
  for (int k = 0; k < 128; ++k) conv[k] = tanh_fast(s[k] + r[k] - t[k]);
  float o = l2b[0];
  for (int j = 0; j < 16; ++j) {
    float a = l1b[j];
    for (int k = 0; k < 128; ++k) a += conv[k] * l1W[k * 16 + j];
    o += a * l2W[j];
  }
  out[b] = o;
}

// ---------------------------------------------------------------- bidirectional GRU
// gh(b,:) = h(b,:) @ Whh depends only on row b => 16 independent single-wave scans
// (dir x batch-row-block). Hidden state lives in an LDS ping-pong buffer (2x8KB per
// wave) to keep VGPR pressure low: A-fragments are built straight from hbuf[cur],
// gate math reads old h from hbuf[cur] and writes new h into hbuf[nxt].
// Same-wave LDS ops are in-order (ISA 11), so no barriers are needed.
__global__ __launch_bounds__(32) void gru_k(const float* gi,        // [2][N][384] (incl bih)
                                            const float* h0,        // [B][128]
                                            const float* bhh2,      // [2][384]
                                            const __bf16* whhT2,    // [2][384][128]
                                            float* out) {           // [N][256]
  const int dir = blockIdx.x >> 3;
  const int wv = blockIdx.x & 7;
  const int lane = threadIdx.x;
  const int colr = lane & 15, rb = (lane >> 4) * 8;
  const int r = lane & 15, half = lane >> 4;
  __shared__ float hbuf[2][16 * 128];
  const float* giD = gi + (size_t)dir * kN * 384;
  const float* bh = bhh2 + dir * 384;
  const __bf16* W = whhT2 + (size_t)dir * 384 * 128;

  // init: rows 16wv..16wv+15 of h0 are one contiguous 8KB run
  const float* h0w = h0 + (size_t)(16 * wv) * 128;
  for (int i = lane; i < 16 * 128; i += 32) hbuf[0][i] = h0w[i];

  for (int s = 0; s < kNPG; ++s) {
    const int t = dir ? (kNPG - 1 - s) : s;
    const float* hc = hbuf[s & 1];
    float* hn = hbuf[(s & 1) ^ 1];
    v16bf afr[4];
#pragma unroll
    for (int f = 0; f < 4; ++f)
      afr[f] = afrag_plain(&hc[r * 128 + 32 * f], half);
#pragma unroll 1
    for (int cb = 0; cb < 8; ++cb) {
      v8f aR = {0.f,0.f,0.f,0.f,0.f,0.f,0.f,0.f};
      v8f aZ = aR, aN = aR;
      const __bf16* wR = W + (size_t)(cb * 16 + r) * 128;
      const __bf16* wZ = W + (size_t)(128 + cb * 16 + r) * 128;
      const __bf16* wN = W + (size_t)(256 + cb * 16 + r) * 128;
#pragma unroll
      for (int f = 0; f < 4; ++f) {
        aR = __builtin_amdgcn_wmma_f32_16x16x32_bf16(false, afr[f], false,
                 bfrag(wR + 32 * f, half), (short)0, aR, false, false);
        aZ = __builtin_amdgcn_wmma_f32_16x16x32_bf16(false, afr[f], false,
                 bfrag(wZ + 32 * f, half), (short)0, aZ, false, false);
        aN = __builtin_amdgcn_wmma_f32_16x16x32_bf16(false, afr[f], false,
                 bfrag(wN + 32 * f, half), (short)0, aN, false, false);
      }
#pragma unroll
      for (int v = 0; v < 8; ++v) {
        int bi = 16 * wv + rb + v;
        size_t node = (size_t)bi * kNPG + t;
        int col = cb * 16 + colr;
        float gr = giD[node * 384 + col];
        float gz = giD[node * 384 + 128 + col];
        float gn = giD[node * 384 + 256 + col];
        float hold = hc[(rb + v) * 128 + col];
        float rg = sigm(gr + aR[v] + bh[col]);
        float zg = sigm(gz + aZ[v] + bh[128 + col]);
        float ng = tanh_fast(gn + rg * (aN[v] + bh[256 + col]));
        float hv = (1.f - zg) * ng + zg * hold;
        hn[(rb + v) * 128 + col] = hv;
        out[node * 256 + dir * 128 + col] = fmaxf(hv, 0.f);   // relu fused
      }
    }
  }
}

// ---------------------------------------------------------------- host
extern "C" void kernel_launch(void* const* d_in, const int* in_sizes, int n_in,
                              void* d_out, int out_size, void* d_ws, size_t ws_size,
                              hipStream_t stream) {
  (void)in_sizes; (void)n_in; (void)out_size; (void)ws_size;
  const float* node_feat = (const float*)d_in[0];
  const int* esrc   = (const int*)d_in[1];
  const int* edst   = (const int*)d_in[2];
  const int* etype  = (const int*)d_in[3];
  const int* glabel = (const int*)d_in[4];
  const int* srcn   = (const int*)d_in[5];
  const int* tgtn   = (const int*)d_in[6];
  const float* P_final_rel = (const float*)d_in[7];
  const float* P_r2eW  = (const float*)d_in[8];
  const float* P_r2eb  = (const float*)d_in[9];
  const float* P_Win   = (const float*)d_in[10];
  const float* P_Wie   = (const float*)d_in[11];
  const float* P_inatt1= (const float*)d_in[12];
  const float* P_inatt2= (const float*)d_in[13];
  const float* P_Whe   = (const float*)d_in[14];
  const float* P_att1  = (const float*)d_in[15];
  const float* P_att2  = (const float*)d_in[16];
  const float* P_Whn   = (const float*)d_in[17];
  const float* P_comm  = (const float*)d_in[18];
  const float* P_Wo    = (const float*)d_in[19];
  const float* P_Wob   = (const float*)d_in[20];
  const float* P_grub  = (const float*)d_in[21];
  const float* P_Wih   = (const float*)d_in[22];
  const float* P_Whh   = (const float*)d_in[23];
  const float* P_bih   = (const float*)d_in[24];
  const float* P_bhh   = (const float*)d_in[25];
  const float* P_l1W   = (const float*)d_in[26];
  const float* P_l1b   = (const float*)d_in[27];
  const float* P_l2W   = (const float*)d_in[28];
  const float* P_l2b   = (const float*)d_in[29];

  // bf16 transposed weight pool (1 MB)
  __bf16* wt = (__bf16*)d_ws;
  size_t wo = 0;
  __bf16* WT_WIN  = wt + wo; wo += 128 * 128;
  __bf16* WT_WIE  = wt + wo; wo += 128 * 384;
  __bf16* WT_R2E  = wt + wo; wo += 128 * 128;
  __bf16* WT_IA1  = wt + wo; wo += 128 * 256;
  __bf16* WT_WHE0 = wt + wo; wo += 128 * 128;
  __bf16* WT_WHE1 = wt + wo; wo += 128 * 128;
  __bf16* WT_A10  = wt + wo; wo += 128 * 256;
  __bf16* WT_A11  = wt + wo; wo += 128 * 256;
  __bf16* WT_WHN0 = wt + wo; wo += 128 * 128;
  __bf16* WT_WHN1 = wt + wo; wo += 128 * 128;
  __bf16* WT_COMM = wt + wo; wo += 128 * 384;
  __bf16* WT_WO   = wt + wo; wo += 128 * 256;
  __bf16* WT_WIH  = wt + wo; wo += 2 * 384 * 128;
  __bf16* WT_WHH  = wt + wo; wo += 2 * 384 * 128;

  // workspace layout (peak ~327 MB; gi/gru_out overlay the dead edge buffers)
  char* base = (char*)d_ws + (1 << 20);
  const size_t szA = (size_t)kN * kH * 4;
  const size_t szE = (size_t)kE * kH * 4;
  const size_t szCat = (size_t)kN * 3 * kH * 4;
  float* Amn  = (float*)(base);                 // message_node [N,128]
  float* tmpN = (float*)(base + szA);           // node tmp / node_hidden
  float* cat  = (float*)(base + 2 * szA);       // [mn|agg_t|agg_s] [N,384]; later msg
  float* msg  = cat;
  float* h0b  = (float*)(base + 2 * szA + szCat);
  char*  bigR = base + 2 * szA + szCat + 65536;
  float* Bie  = (float*)(bigR);                 // input_edge [E,128]
  float* Cme  = (float*)(bigR + szE);           // message_edge [E,128]
  float* Dtr  = (float*)(bigR + 2 * szE);       // tgt_rel_h [E,128]
  float* tmpE = (float*)(bigR + 3 * szE);       // att hidden [E,128]
  float* attb = (float*)(bigR + 4 * szE);       // att [E]
  float* gi   = (float*)(bigR);                 // overlay: [2][N][384]
  float* gout = (float*)(bigR + 2 * (size_t)kN * 384 * 4);  // [N,256]
  float* nh   = tmpN;

  auto wc = [&](const float* src, __bf16* dst, int K, int Nc) {
    int tot = K * Nc;
    wconv_k<<<dim3((tot + 255) / 256), dim3(256), 0, stream>>>(src, dst, K, Nc);
  };
  wc(P_Win, WT_WIN, 128, 128);
  wc(P_Wie, WT_WIE, 384, 128);
  wc(P_r2eW, WT_R2E, 128, 128);
  wc(P_inatt1, WT_IA1, 256, 128);
  wc(P_Whe, WT_WHE0, 128, 128);             wc(P_Whe + 128 * 128, WT_WHE1, 128, 128);
  wc(P_att1, WT_A10, 256, 128);             wc(P_att1 + 256 * 128, WT_A11, 256, 128);
  wc(P_Whn, WT_WHN0, 128, 128);             wc(P_Whn + 128 * 128, WT_WHN1, 128, 128);
  wc(P_comm, WT_COMM, 384, 128);
  wc(P_Wo, WT_WO, 256, 128);
  wc(P_Wih, WT_WIH, 128, 384);              wc(P_Wih + 128 * 384, WT_WIH + 384 * 128, 128, 384);
  wc(P_Whh, WT_WHH, 128, 384);              wc(P_Whh + 128 * 384, WT_WHH + 384 * 128, 128, 384);

  auto gemm = [&](const GemmArgs& g) {
    dim3 grd(g.M / 128, g.Nc / 32), blk(256);
    switch (g.mode) {
      case 0: gemm_k<0><<<grd, blk, 0, stream>>>(g); break;
      case 1: gemm_k<1><<<grd, blk, 0, stream>>>(g); break;
      case 2: gemm_k<2><<<grd, blk, 0, stream>>>(g); break;
      case 3: gemm_k<3><<<grd, blk, 0, stream>>>(g); break;
      case 4: gemm_k<4><<<grd, blk, 0, stream>>>(g); break;
      default: gemm_k<5><<<grd, blk, 0, stream>>>(g); break;
    }
  };
  GemmArgs G{};
  G.rel = P_final_rel; G.esrc = esrc; G.edst = edst; G.etype = etype;
  G.glabel = glabel; G.spe = srcn; G.dpe = tgtn; G.trh = Dtr;

  // 1. input_node = relu(node_feat @ W_i_node)  -> Amn (== message_node)
  { GemmArgs g = G; g.mode = 0; g.M = kN; g.K = 128; g.Nc = 128; g.A = node_feat;
    g.lda = 128; g.W = WT_WIN; g.act = 1; g.out = Amn; g.ldo = 128; gemm(g); }
  // 2. input_edge = relu([nf[src]|rel[type]|nf[dst]] @ W_i_edge)  -> Bie
  { GemmArgs g = G; g.mode = 1; g.M = kE; g.K = 384; g.Nc = 128; g.nodef = node_feat;
    g.W = WT_WIE; g.act = 1; g.out = Bie; g.ldo = 128; gemm(g); }
  // 3. tgt_rel_h = rel[glabel[g(e)]] @ rel2edge_W + b  -> Dtr
  { GemmArgs g = G; g.mode = 2; g.M = kE; g.K = 128; g.Nc = 128;
    g.W = WT_R2E; g.bias = P_r2eb; g.out = Dtr; g.ldo = 128; gemm(g); }
  // 4. init attention hidden = relu([ge|input_edge] @ in_att1) -> tmpE; att = sig(.@in_att2)
  { GemmArgs g = G; g.mode = 4; g.M = kE; g.K = 256; g.Nc = 128; g.nodef = Amn;
    g.medge = Bie; g.W = WT_IA1; g.act = 1; g.out = tmpE; g.ldo = 128; gemm(g); }
  attdot_k<<<dim3(kE / 256), dim3(256), 0, stream>>>(tmpE, P_inatt2, attb);

  const __bf16* WHEd[2] = {WT_WHE0, WT_WHE1};
  const __bf16* A1d[2] = {WT_A10, WT_A11};
  const __bf16* WHNd[2] = {WT_WHN0, WT_WHN1};
  for (int d = 0; d < 2; ++d) {
    scale_k<<<dim3(kE * kH / 256), dim3(256), 0, stream>>>(d == 0 ? Bie : Cme, attb, Cme);
    catprep_k<<<dim3(kN * kH / 256), dim3(256), 0, stream>>>(Amn, cat);
    scatter_k<<<dim3(kE), dim3(128), 0, stream>>>(Cme, edst, esrc, cat);
    { GemmArgs g = G; g.mode = 0; g.M = kN; g.K = 384; g.Nc = 128; g.A = cat; g.lda = 384;
      g.W = WT_COMM; g.act = 1; g.out = tmpN; g.ldo = 128; gemm(g); }          // comm_mlp
    { GemmArgs g = G; g.mode = 0; g.M = kN; g.K = 128; g.Nc = 128; g.A = tmpN; g.lda = 128;
      g.W = WHNd[d]; g.act = 1; g.out = Amn; g.ldo = 128; gemm(g); }           // W_h_node
    { GemmArgs g = G; g.mode = 3; g.M = kE; g.K = 128; g.Nc = 128; g.nodef = Amn;
      g.W = WHEd[d]; g.resid = Bie; g.ldr = 128; g.act = 1; g.out = Cme; g.ldo = 128;
      gemm(g); }                                                               // W_h_edge
    { GemmArgs g = G; g.mode = 5; g.M = kE; g.K = 256; g.Nc = 128; g.medge = Cme;
      g.W = A1d[d]; g.act = 1; g.out = tmpE; g.ldo = 128; gemm(g); }           // att1
    attdot_k<<<dim3(kE / 256), dim3(256), 0, stream>>>(tmpE, P_att2 + d * 128, attb);
  }
  // final aggregation + comm_mlp -> Amn
  scale_k<<<dim3(kE * kH / 256), dim3(256), 0, stream>>>(Cme, attb, Cme);
  catprep_k<<<dim3(kN * kH / 256), dim3(256), 0, stream>>>(Amn, cat);
  scatter_k<<<dim3(kE), dim3(128), 0, stream>>>(Cme, edst, esrc, cat);
  { GemmArgs g = G; g.mode = 0; g.M = kN; g.K = 384; g.Nc = 128; g.A = cat; g.lda = 384;
    g.W = WT_COMM; g.act = 1; g.out = Amn; g.ldo = 128; gemm(g); }

  // GRU: msg, h0, gi (batched x@Wih+bih), then 16 independent wave scans
  msg_k<<<dim3(kN * kH / 256), dim3(256), 0, stream>>>(Amn, P_grub, msg);
  h0max_k<<<dim3(kB * kH / 256), dim3(256), 0, stream>>>(msg, h0b);
  for (int d = 0; d < 2; ++d) {
    GemmArgs g = G; g.mode = 0; g.M = kN; g.K = 128; g.Nc = 384; g.A = msg; g.lda = 128;
    g.W = WT_WIH + (size_t)d * 384 * 128; g.bias = P_bih + d * 384; g.act = 0;
    g.out = gi + (size_t)d * kN * 384; g.ldo = 384; gemm(g);
  }
  gru_k<<<dim3(16), dim3(32), 0, stream>>>(gi, h0b, P_bhh, WT_WHH, gout);

  // node_hidden = relu(gru_out @ W_o + b) -> nh; then head -> d_out [B]
  { GemmArgs g = G; g.mode = 0; g.M = kN; g.K = 256; g.Nc = 128; g.A = gout; g.lda = 256;
    g.W = WT_WO; g.bias = P_Wob; g.act = 1; g.out = nh; g.ldo = 128; gemm(g); }
  head_k<<<dim3(1), dim3(128), 0, stream>>>(nh, P_final_rel, srcn, tgtn, glabel,
                                            P_l1W, P_l1b, P_l2W, P_l2b, (float*)d_out);
}
